// RQVAE_31903017075081
// MI455X (gfx1250) — compile-verified, matrix operand order checked by reading, not run on hardware
//
#include <hip/hip_runtime.h>
#include <stdint.h>

// Problem sizes (fixed by the reference)
#define N_ROWS 16384
#define DIN    2048
#define HDIM   4096
#define CDIM   256
#define KCB    256

typedef __attribute__((ext_vector_type(16))) __bf16 v16bf;
typedef __attribute__((ext_vector_type(8)))  float  v8f;

__device__ __forceinline__ unsigned short f2bf(float f) {
    // round-to-nearest-even f32 -> bf16
    unsigned u = __float_as_uint(f);
    u += 0x7FFFu + ((u >> 16) & 1u);
    return (unsigned short)(u >> 16);
}

union FragAB {
    v16bf v;
    uint4 q[2];
    unsigned short us[16];
};
union FragC {
    v8f  v;
    float f[8];
};

// ---------------------------------------------------------------------------
// f32 -> bf16 elementwise convert (for activations entering a WMMA GEMM)
// ---------------------------------------------------------------------------
__global__ void cvt_bf16_kernel(const float* __restrict__ src,
                                unsigned short* __restrict__ dst, int n) {
    int i = blockIdx.x * blockDim.x + threadIdx.x;
    if (i < n) dst[i] = f2bf(src[i]);
}

// ---------------------------------------------------------------------------
// Pack a weight matrix into the WMMA B-fragment layout.
//   dst layout: [kb][nb][lane(0..31)][j(0..15)] bf16, tile = 32(K) x 16(N)
//   lane l (l<16):  col = nb*16 + l,  K = kb*32 + {0..7 | 16..23}
//   lane l (l>=16): col = nb*16+l-16, K = kb*32 + {8..15 | 24..31}
// transposed==0: src is [Kdim x Nout] row-major (B = W, for z @ W)
// transposed==1: src is [Nout x Kdim] row-major (B = cb^T, for z @ cb^T)
// ---------------------------------------------------------------------------
__global__ void pack_b_kernel(const float* __restrict__ src,
                              unsigned short* __restrict__ dst,
                              int Kdim, int Nout, int transposed) {
    int p = blockIdx.x * blockDim.x + threadIdx.x;
    int total = Kdim * Nout;
    if (p >= total) return;
    int within = p & 511;
    int tile   = p >> 9;
    int lane   = within >> 4;
    int j      = within & 15;
    int tilesN = Nout >> 4;
    int kb  = tile / tilesN;
    int nb  = tile - kb * tilesN;
    int col = (nb << 4) + (lane & 15);
    int kk  = (kb << 5) + ((j < 8) ? 0 : 16) + ((lane >> 4) << 3) + (j & 7);
    float v = transposed ? src[(size_t)col * Kdim + kk]
                         : src[(size_t)kk * Nout + col];
    dst[p] = f2bf(v);
}

// ---------------------------------------------------------------------------
// Codebook squared norms: ||cb[k]||^2 for 3 codebooks (blockIdx = which book)
// ---------------------------------------------------------------------------
__global__ void cbnorm_kernel(const float* __restrict__ cb1,
                              const float* __restrict__ cb2,
                              const float* __restrict__ cb3,
                              float* __restrict__ out) {
    const float* cb = (blockIdx.x == 0) ? cb1 : (blockIdx.x == 1) ? cb2 : cb3;
    int k = threadIdx.x;
    float s = 0.f;
    for (int c = 0; c < CDIM; ++c) {
        float v = cb[(size_t)k * CDIM + c];
        s += v * v;
    }
    out[blockIdx.x * KCB + k] = s;
}

// ---------------------------------------------------------------------------
// bf16 WMMA GEMM: C[M,Nout] = act(A[M,Kdim] @ B + bias)
// Each wave computes a 32x64 strip: 2 A fragments x 4 B fragments ->
// 8 WMMAs per 32-deep K step against 12 b128 loads (B reused across both
// M tiles), keeping the XDL pipe (not VMEM issue) the limiter.
// act: 0=none, 1=relu, 2=sigmoid.  Cf (f32) and Cb (bf16) each optional.
// ---------------------------------------------------------------------------
__global__ __launch_bounds__(256) void gemm_bf16_wmma(
    const unsigned short* __restrict__ A,   // [M x Kdim] row-major bf16
    const unsigned short* __restrict__ Bp,  // packed B fragments
    const float* __restrict__ bias,         // [Nout] or null
    float* __restrict__ Cf,                 // [M x Nout] f32 or null
    unsigned short* __restrict__ Cb,        // [M x Nout] bf16 or null
    int M, int Kdim, int Nout, int act) {
    const int wave = (blockIdx.x * blockDim.x + threadIdx.x) >> 5;
    const int lane = threadIdx.x & 31;
    const int stripsN = Nout >> 6;
    const int totalWaves = (M >> 5) * stripsN;   // 32-row x 64-col strips
    if (wave >= totalWaves) return;
    const int m0 = (wave / stripsN) << 5;
    const int n0 = (wave % stripsN) << 6;

    const int half = lane >> 4;                       // 0 or 1
    const unsigned short* aBase0 =
        A + (size_t)(m0 + (lane & 15)) * Kdim + (half << 3);
    const unsigned short* aBase1 = aBase0 + (size_t)16 * Kdim;
    const int tilesN = Nout >> 4;
    const unsigned short* bBase =
        Bp + ((size_t)(n0 >> 4) << 9) + ((size_t)lane << 4);
    const size_t bKStride = (size_t)tilesN << 9;      // shorts per kb step

    FragC acc[8];
#pragma unroll
    for (int t = 0; t < 8; ++t)
#pragma unroll
        for (int i = 0; i < 8; ++i) acc[t].f[i] = 0.f;

    const int kbMax = Kdim >> 5;
    for (int kb = 0; kb < kbMax; ++kb) {
        const unsigned short* ap0 = aBase0 + (kb << 5);
        const unsigned short* ap1 = aBase1 + (kb << 5);
        const unsigned short* bp  = bBase + kb * bKStride;
        // prefetch next iteration (speculative; gfx1250 global_prefetch_b8)
        __builtin_prefetch(ap0 + 32, 0, 1);
        __builtin_prefetch(ap1 + 32, 0, 1);
        __builtin_prefetch(bp + bKStride, 0, 1);

        FragAB a0, a1;
        a0.q[0] = *reinterpret_cast<const uint4*>(ap0);
        a0.q[1] = *reinterpret_cast<const uint4*>(ap0 + 16);
        a1.q[0] = *reinterpret_cast<const uint4*>(ap1);
        a1.q[1] = *reinterpret_cast<const uint4*>(ap1 + 16);

#pragma unroll
        for (int nt = 0; nt < 4; ++nt) {
            FragAB b;
            b.q[0] = *reinterpret_cast<const uint4*>(bp + (nt << 9));
            b.q[1] = *reinterpret_cast<const uint4*>(bp + (nt << 9) + 8);
            acc[nt].v = __builtin_amdgcn_wmma_f32_16x16x32_bf16(
                false, a0.v, false, b.v, (short)0, acc[nt].v, false, false);
            acc[4 + nt].v = __builtin_amdgcn_wmma_f32_16x16x32_bf16(
                false, a1.v, false, b.v, (short)0, acc[4 + nt].v, false, false);
        }
    }

    // Write out: acc VGPR r: lanes0-15 -> M=rowBase+r, lanes16-31 -> +8
#pragma unroll
    for (int mt = 0; mt < 2; ++mt) {
        const int rowBase = m0 + (mt << 4) + (half << 3);
#pragma unroll
        for (int nt = 0; nt < 4; ++nt) {
            const int col = n0 + (nt << 4) + (lane & 15);
            const float bv = bias ? bias[col] : 0.f;
            FragC& a = acc[mt * 4 + nt];
#pragma unroll
            for (int r = 0; r < 8; ++r) {
                float v = a.f[r] + bv;
                if (act == 1) v = v > 0.f ? v : 0.f;
                else if (act == 2) v = 1.f / (1.f + __expf(-v));
                const size_t idx = (size_t)(rowBase + r) * Nout + col;
                if (Cf) Cf[idx] = v;
                if (Cb) Cb[idx] = f2bf(v);
            }
        }
    }
}

// ---------------------------------------------------------------------------
// VQ stage reduce: one wave per row. argmin_k (||cb_k||^2 - 2 S[n,k]),
// gather zq, residual, qsum accumulation, bf16 staging for next GEMM.
// ---------------------------------------------------------------------------
__global__ __launch_bounds__(256) void vq_reduce_kernel(
    const float* __restrict__ S,          // [N x K] = z . cb^T
    const float* __restrict__ cb,         // [K x C] f32
    const float* __restrict__ cbnorm,     // [K]
    const float* __restrict__ zeIn,       // [N x C] f32 (current residual)
    float* __restrict__ nearestOut,       // [N] (index stored as float)
    float* __restrict__ zqOut,            // [N x C]
    float* __restrict__ zeNextOut,        // [N x C] or null
    unsigned short* __restrict__ zebNext, // [N x C] bf16 or null
    float* __restrict__ qsum,             // [N x C]
    unsigned short* __restrict__ decinb,  // [N x C] bf16 or null (stage 3)
    int stage) {
    const int row  = (blockIdx.x * blockDim.x + threadIdx.x) >> 5;
    const int lane = threadIdx.x & 31;
    if (row >= N_ROWS) return;

    const float* s = S + (size_t)row * KCB;
    float bestd = 3.4e38f;
    int bestk = 0;
#pragma unroll
    for (int j = 0; j < 8; ++j) {
        int k = lane + (j << 5);
        float d = cbnorm[k] - 2.f * s[k];
        if (d < bestd) { bestd = d; bestk = k; }
    }
    // wave32 butterfly argmin (first-index tie-break)
    for (int off = 16; off > 0; off >>= 1) {
        float od = __shfl_xor(bestd, off, 32);
        int   ok = __shfl_xor(bestk, off, 32);
        if (od < bestd || (od == bestd && ok < bestk)) { bestd = od; bestk = ok; }
    }
    if (lane == 0) nearestOut[row] = (float)bestk;

    const float* cbr = cb + (size_t)bestk * CDIM;
#pragma unroll
    for (int j = 0; j < 8; ++j) {
        int c = lane + (j << 5);
        size_t idx = (size_t)row * CDIM + c;
        float q = cbr[c];
        float z = zeIn[idx];
        float r = z - q;
        zqOut[idx] = q;
        if (zeNextOut) zeNextOut[idx] = r;
        if (zebNext)   zebNext[idx]   = f2bf(r);
        float qs = (stage == 1) ? q : (qsum[idx] + q);
        qsum[idx] = qs;
        if (decinb) decinb[idx] = f2bf(qs);
    }
}

// ---------------------------------------------------------------------------
// Host-side orchestration
// ---------------------------------------------------------------------------
extern "C" void kernel_launch(void* const* d_in, const int* in_sizes, int n_in,
                              void* d_out, int out_size, void* d_ws,
                              size_t ws_size, hipStream_t stream) {
    const float* x    = (const float*)d_in[0];
    const float* W_e1 = (const float*)d_in[1];
    const float* b_e1 = (const float*)d_in[2];
    const float* W_e2 = (const float*)d_in[3];
    const float* b_e2 = (const float*)d_in[4];
    const float* cb1  = (const float*)d_in[5];
    const float* cb2  = (const float*)d_in[6];
    const float* cb3  = (const float*)d_in[7];
    const float* W_d1 = (const float*)d_in[8];
    const float* b_d1 = (const float*)d_in[9];
    const float* W_d2 = (const float*)d_in[10];
    const float* b_d2 = (const float*)d_in[11];

    // ---- workspace carve-out (bytes, 256B aligned) ----
    char* ws = (char*)d_ws;
    size_t off = 0;
    auto carve = [&](size_t bytes) -> void* {
        void* p = ws + off;
        off = (off + bytes + 255) & ~(size_t)255;
        return p;
    };
    unsigned short* xb    = (unsigned short*)carve((size_t)N_ROWS * DIN * 2);
    unsigned short* h1b   = (unsigned short*)carve((size_t)N_ROWS * HDIM * 2);
    unsigned short* zeb   = (unsigned short*)carve((size_t)N_ROWS * CDIM * 2);
    unsigned short* dinb  = (unsigned short*)carve((size_t)N_ROWS * CDIM * 2);
    unsigned short* hdecb = (unsigned short*)carve((size_t)N_ROWS * HDIM * 2);
    float*          Ssc   = (float*)carve((size_t)N_ROWS * KCB * 4);
    float*          qsum  = (float*)carve((size_t)N_ROWS * CDIM * 4);
    unsigned short* We1p  = (unsigned short*)carve((size_t)DIN * HDIM * 2);
    unsigned short* We2p  = (unsigned short*)carve((size_t)HDIM * CDIM * 2);
    unsigned short* Wd1p  = (unsigned short*)carve((size_t)CDIM * HDIM * 2);
    unsigned short* Wd2p  = (unsigned short*)carve((size_t)HDIM * DIN * 2);
    unsigned short* cbp1  = (unsigned short*)carve((size_t)KCB * CDIM * 2);
    unsigned short* cbp2  = (unsigned short*)carve((size_t)KCB * CDIM * 2);
    unsigned short* cbp3  = (unsigned short*)carve((size_t)KCB * CDIM * 2);
    float*          cbn   = (float*)carve(3 * KCB * 4);

    // ---- output layout (f32, concatenated in return order) ----
    float* out  = (float*)d_out;
    float* xhat = out;
    float* ze1  = xhat + (size_t)N_ROWS * DIN;
    float* ze2  = ze1 + (size_t)N_ROWS * CDIM;
    float* ze3  = ze2 + (size_t)N_ROWS * CDIM;
    float* zq1  = ze3 + (size_t)N_ROWS * CDIM;
    float* zq2  = zq1 + (size_t)N_ROWS * CDIM;
    float* zq3  = zq2 + (size_t)N_ROWS * CDIM;
    float* nr1  = zq3 + (size_t)N_ROWS * CDIM;
    float* nr2  = nr1 + N_ROWS;
    float* nr3  = nr2 + N_ROWS;

    // ---- prep: bf16 conversion + weight packing ----
    {
        int n = N_ROWS * DIN;
        cvt_bf16_kernel<<<(n + 255) / 256, 256, 0, stream>>>(x, xb, n);
    }
    auto pack = [&](const float* src, unsigned short* dst, int Kd, int No, int tr) {
        int total = Kd * No;
        pack_b_kernel<<<(total + 255) / 256, 256, 0, stream>>>(src, dst, Kd, No, tr);
    };
    pack(W_e1, We1p, DIN, HDIM, 0);
    pack(W_e2, We2p, HDIM, CDIM, 0);
    pack(W_d1, Wd1p, CDIM, HDIM, 0);
    pack(W_d2, Wd2p, HDIM, DIN, 0);
    pack(cb1, cbp1, CDIM, KCB, 1);   // B = cb^T
    pack(cb2, cbp2, CDIM, KCB, 1);
    pack(cb3, cbp3, CDIM, KCB, 1);
    cbnorm_kernel<<<3, 256, 0, stream>>>(cb1, cb2, cb3, cbn);

    auto gemm = [&](const unsigned short* A, const unsigned short* Bp,
                    const float* bias, float* Cf, unsigned short* Cb,
                    int M, int Kd, int No, int act) {
        int waves = (M / 32) * (No / 64);
        int blocks = (waves + 7) / 8;
        gemm_bf16_wmma<<<blocks, 256, 0, stream>>>(A, Bp, bias, Cf, Cb,
                                                   M, Kd, No, act);
    };

    // ---- encoder ----
    gemm(xb, We1p, b_e1, nullptr, h1b, N_ROWS, DIN, HDIM, /*relu*/ 1);
    gemm(h1b, We2p, b_e2, ze1, zeb, N_ROWS, HDIM, CDIM, 0);

    // ---- residual VQ (3 stages) ----
    const int rblocks = N_ROWS / 8;  // one wave per row, 8 waves/block
    gemm(zeb, cbp1, nullptr, Ssc, nullptr, N_ROWS, CDIM, KCB, 0);
    vq_reduce_kernel<<<rblocks, 256, 0, stream>>>(
        Ssc, cb1, cbn + 0 * KCB, ze1, nr1, zq1, ze2, zeb, qsum, nullptr, 1);

    gemm(zeb, cbp2, nullptr, Ssc, nullptr, N_ROWS, CDIM, KCB, 0);
    vq_reduce_kernel<<<rblocks, 256, 0, stream>>>(
        Ssc, cb2, cbn + 1 * KCB, ze2, nr2, zq2, ze3, zeb, qsum, nullptr, 2);

    gemm(zeb, cbp3, nullptr, Ssc, nullptr, N_ROWS, CDIM, KCB, 0);
    vq_reduce_kernel<<<rblocks, 256, 0, stream>>>(
        Ssc, cb3, cbn + 2 * KCB, ze3, nr3, zq3, nullptr, nullptr, qsum, dinb, 3);

    // ---- decoder (input = zq1+zq2+zq3, straight-through forward value) ----
    gemm(dinb, Wd1p, b_d1, nullptr, hdecb, N_ROWS, CDIM, HDIM, /*sigmoid*/ 2);
    gemm(hdecb, Wd2p, b_d2, xhat, nullptr, N_ROWS, HDIM, DIN, 0);
}
